// Block_65257733096091
// MI455X (gfx1250) — 8-bit matrix operand order repaired and checked by reading, compiled, not run on hardware
//
#include <hip/hip_runtime.h>
#include <stdint.h>

#define C_IN    32
#define C_OUT   32
#define PLANES  192
#define HW      12544
#define NBATCH  32
#define NPIX    (NBATCH * HW)
#define QMAXF   7.0f

#define OFF_W1Q 64
#define OFF_W2Q (OFF_W1Q + PLANES * C_IN)
#define OFF_W3Q 7936
#define OFF_A1  16384
#define OFF_A2  (OFF_A1 + (size_t)NPIX * PLANES * 2)

typedef int v8i __attribute__((ext_vector_type(8)));

__device__ __forceinline__ int quant4(float v, float inv) {
  int q = (int)rintf(v * inv);
  return q < -8 ? -8 : (q > 7 ? 7 : q);
}
__device__ __forceinline__ float mkscale(unsigned int bits) {
  return fmaxf(__uint_as_float(bits) / QMAXF, 1e-8f);
}

__global__ void k_init(unsigned int* wsh) {
  if (threadIdx.x < 6) wsh[threadIdx.x] = 0u;
}

__global__ void k_amax(const float4* __restrict__ p, int n4, unsigned int* dst) {
  __shared__ unsigned int red[256];
  unsigned int m = 0u;
  for (int i = blockIdx.x * blockDim.x + threadIdx.x; i < n4;
       i += gridDim.x * blockDim.x) {
    float4 v = p[i];
    m = max(m, __float_as_uint(fabsf(v.x)));
    m = max(m, __float_as_uint(fabsf(v.y)));
    m = max(m, __float_as_uint(fabsf(v.z)));
    m = max(m, __float_as_uint(fabsf(v.w)));
  }
  red[threadIdx.x] = m;
  __syncthreads();
  for (int s = 128; s > 0; s >>= 1) {
    if ((int)threadIdx.x < s) red[threadIdx.x] = max(red[threadIdx.x], red[threadIdx.x + s]);
    __syncthreads();
  }
  if (threadIdx.x == 0) atomicMax(dst, red[0]);
}

__global__ void k_quantw(const float* __restrict__ w1, const float* __restrict__ w2,
                         const float* __restrict__ w3, const unsigned int* __restrict__ wsh,
                         signed char* __restrict__ ws) {
  float i1 = 1.0f / mkscale(wsh[1]);
  float i2 = 1.0f / mkscale(wsh[2]);
  float i3 = 1.0f / mkscale(wsh[3]);
  int stride = gridDim.x * blockDim.x;
  int t = blockIdx.x * blockDim.x + threadIdx.x;
  for (int i = t; i < PLANES * C_IN; i += stride) ws[OFF_W1Q + i] = (signed char)quant4(w1[i], i1);
  for (int i = t; i < PLANES * 9;   i += stride) ws[OFF_W2Q + i] = (signed char)quant4(w2[i], i2);
  for (int i = t; i < C_OUT * PLANES; i += stride) ws[OFF_W3Q + i] = (signed char)quant4(w3[i], i3);
}

__global__ void k_conv1(const float* __restrict__ x, const signed char* __restrict__ wq,
                        const unsigned int* __restrict__ wsh,
                        short* __restrict__ a1, int* __restrict__ amax1) {
  __shared__ __align__(16) signed char s_w[PLANES * C_IN];
  __shared__ __align__(16) signed char s_x[128 * C_IN];
  __shared__ int s_red[256];

  const int tid = threadIdx.x;
  const int pixbase = blockIdx.x * 128;
  const int n  = pixbase / HW;
  const int hw = pixbase % HW;

  const float invx = 1.0f / mkscale(wsh[0]);

  for (int i = tid; i < PLANES * C_IN / 4; i += 256)
    ((int*)s_w)[i] = ((const int*)wq)[i];

  const float* xb = x + (size_t)n * C_IN * HW + hw;
  for (int j = 0; j < 4; ++j) {
    int idx = j * 1024 + tid * 4;
    int ic = idx >> 7;
    int px = idx & 127;
    float4 v = *(const float4*)(xb + (size_t)ic * HW + px);
    s_x[(px + 0) * C_IN + ic] = (signed char)quant4(v.x, invx);
    s_x[(px + 1) * C_IN + ic] = (signed char)quant4(v.y, invx);
    s_x[(px + 2) * C_IN + ic] = (signed char)quant4(v.z, invx);
    s_x[(px + 3) * C_IN + ic] = (signed char)quant4(v.w, invx);
  }
  __syncthreads();

  const int lane = tid & 31;
  const int wave = tid >> 5;
  const int nn = lane & 15;
  const int hi = lane >> 4;
  const int pb = wave * 16;

  v8i B = {};
  {
    const signed char* bp = &s_x[(pb + nn) * C_IN + hi * 16];
    int4 b0 = *(const int4*)bp;
    B[0] = b0.x; B[1] = b0.y; B[2] = b0.z; B[3] = b0.w;
  }

  int tmax = 0;
  short* orow = a1 + (size_t)(pixbase + pb + nn) * PLANES + hi * 8;

#pragma unroll 4
  for (int ot = 0; ot < PLANES / 16; ++ot) {
    const signed char* ap = &s_w[(ot * 16 + nn) * C_IN + hi * 16];
    int2 a0 = *(const int2*)ap;
    int2 a1v = *(const int2*)(ap + 8);
    v8i A = {};
    A[0] = a0.x; A[1] = a0.y; A[2] = a1v.x; A[3] = a1v.y;
    v8i Cz = {};
    v8i D = __builtin_amdgcn_wmma_i32_16x16x64_iu8(true, A, true, B, Cz, false, false);
    int acc[8];
#pragma unroll
    for (int r = 0; r < 8; ++r) {
      int a = D[r];
      acc[r] = a > 0 ? a : 0;
      tmax = max(tmax, acc[r]);
    }
    int4 st;
    st.x = acc[0] | (acc[1] << 16);
    st.y = acc[2] | (acc[3] << 16);
    st.z = acc[4] | (acc[5] << 16);
    st.w = acc[6] | (acc[7] << 16);
    *(int4*)(orow + ot * 16) = st;
  }

  s_red[tid] = tmax;
  __syncthreads();
  for (int s = 128; s > 0; s >>= 1) {
    if (tid < s) s_red[tid] = max(s_red[tid], s_red[tid + s]);
    __syncthreads();
  }
  if (tid == 0) atomicMax(amax1, s_red[0]);
}

__global__ void k_dw(const short* __restrict__ a1, const signed char* __restrict__ w2q,
                     const unsigned int* __restrict__ wsh,
                     short* __restrict__ a2, int* __restrict__ amax2) {
  __shared__ __align__(16) signed char s_q[10 * 10 * PLANES];
  __shared__ __align__(4)  signed char s_w2[PLANES * 9];
  __shared__ int s_red[256];

  const int tid = threadIdx.x;
  const int n = blockIdx.x / 196;
  const int t = blockIdx.x % 196;
  const int th0 = (t / 14) * 8;
  const int tw0 = (t % 14) * 8;

  float k1 = mkscale(wsh[0]) * mkscale(wsh[1]);
  int a1m = ((const int*)wsh)[4];
  float s2 = fmaxf((float)a1m * k1 / QMAXF, 1e-8f);
  float inv = k1 / s2;

  for (int i = tid; i < PLANES * 9 / 4; i += 256)
    ((int*)s_w2)[i] = ((const int*)w2q)[i];

  const short* ab = a1 + (size_t)n * HW * PLANES;
  for (int i = tid; i < 100 * PLANES / 4; i += 256) {
    int s4 = i * 4;
    int pix = s4 / PLANES;
    int ch  = s4 % PLANES;
    int h = th0 + pix / 10 - 1;
    int w = tw0 + pix % 10 - 1;
    int pk = 0;
    if (h >= 0 && h < 112 && w >= 0 && w < 112) {
      int2 d = *(const int2*)(ab + (size_t)(h * 112 + w) * PLANES + ch);
      int q0 = quant4((float)(short)(d.x),       inv);
      int q1 = quant4((float)(short)(d.x >> 16), inv);
      int q2 = quant4((float)(short)(d.y),       inv);
      int q3 = quant4((float)(short)(d.y >> 16), inv);
      pk = (q0 & 255) | ((q1 & 255) << 8) | ((q2 & 255) << 16) | (q3 << 24);
    }
    ((int*)s_q)[i] = pk;
  }
  __syncthreads();

  int tmax = 0;
  short* ob = a2 + (size_t)n * HW * PLANES;
  for (int i = tid; i < 8 * 8 * PLANES; i += 256) {
    int ch = i % PLANES;
    int px = i / PLANES;
    int py = px >> 3, pxx = px & 7;
    const signed char* c  = &s_q[(py * 10 + pxx) * PLANES + ch];
    const signed char* wv = &s_w2[ch * 9];
    int acc = wv[0]*c[0]           + wv[1]*c[PLANES]       + wv[2]*c[2*PLANES]
            + wv[3]*c[10*PLANES]   + wv[4]*c[11*PLANES]    + wv[5]*c[12*PLANES]
            + wv[6]*c[20*PLANES]   + wv[7]*c[21*PLANES]    + wv[8]*c[22*PLANES];
    acc = acc > 0 ? acc : 0;
    tmax = max(tmax, acc);
    ob[(size_t)((th0 + py) * 112 + tw0 + pxx) * PLANES + ch] = (short)acc;
  }

  s_red[tid] = tmax;
  __syncthreads();
  for (int s = 128; s > 0; s >>= 1) {
    if (tid < s) s_red[tid] = max(s_red[tid], s_red[tid + s]);
    __syncthreads();
  }
  if (tid == 0) atomicMax(amax2, s_red[0]);
}

__global__ void k_conv3(const short* __restrict__ a2, const signed char* __restrict__ w3q,
                        const unsigned int* __restrict__ wsh,
                        const float* __restrict__ x, float* __restrict__ out) {
  __shared__ __align__(16) signed char s_w[C_OUT * PLANES];
  __shared__ __align__(16) signed char s_q[128 * PLANES];
  __shared__ __align__(16) float s_out[C_OUT * 128];

  const int tid = threadIdx.x;
  const int pixbase = blockIdx.x * 128;
  const int n  = pixbase / HW;
  const int hw = pixbase % HW;

  float sx = mkscale(wsh[0]), sw1 = mkscale(wsh[1]);
  float sw2 = mkscale(wsh[2]), sw3 = mkscale(wsh[3]);
  int a1m = ((const int*)wsh)[4];
  int a2m = ((const int*)wsh)[5];
  float s2 = fmaxf((float)a1m * sx * sw1 / QMAXF, 1e-8f);
  float s3 = fmaxf((float)a2m * s2 * sw2 / QMAXF, 1e-8f);
  float invq = (s2 * sw2) / s3;
  float deq  = s3 * sw3;

  for (int i = tid; i < C_OUT * PLANES / 4; i += 256)
    ((int*)s_w)[i] = ((const int*)w3q)[i];

  const short* ab = a2 + (size_t)pixbase * PLANES;
  for (int k = tid; k < 128 * PLANES / 8; k += 256) {
    int4 d = ((const int4*)ab)[k];
    int q0 = quant4((float)(short)(d.x),       invq);
    int q1 = quant4((float)(short)(d.x >> 16), invq);
    int q2 = quant4((float)(short)(d.y),       invq);
    int q3 = quant4((float)(short)(d.y >> 16), invq);
    int q4 = quant4((float)(short)(d.z),       invq);
    int q5 = quant4((float)(short)(d.z >> 16), invq);
    int q6 = quant4((float)(short)(d.w),       invq);
    int q7 = quant4((float)(short)(d.w >> 16), invq);
    int2 pk;
    pk.x = (q0 & 255) | ((q1 & 255) << 8) | ((q2 & 255) << 16) | (q3 << 24);
    pk.y = (q4 & 255) | ((q5 & 255) << 8) | ((q6 & 255) << 16) | (q7 << 24);
    ((int2*)s_q)[k] = pk;
  }
  __syncthreads();

  const int lane = tid & 31;
  const int wave = tid >> 5;
  const int nn = lane & 15;
  const int hi = lane >> 4;
  const int pb = wave * 16;

#pragma unroll
  for (int ot = 0; ot < 2; ++ot) {
    v8i Cacc = {};
#pragma unroll
    for (int kt = 0; kt < 3; ++kt) {
      const signed char* bp = &s_q[(pb + nn) * PLANES + kt * 64 + hi * 16];
      int4 b0 = *(const int4*)bp;
      int4 b1 = *(const int4*)(bp + 32);
      v8i B;
      B[0] = b0.x; B[1] = b0.y; B[2] = b0.z; B[3] = b0.w;
      B[4] = b1.x; B[5] = b1.y; B[6] = b1.z; B[7] = b1.w;

      const signed char* ap = &s_w[(ot * 16 + nn) * PLANES + kt * 64 + hi * 16];
      int2 a0 = *(const int2*)ap;
      int2 a1v = *(const int2*)(ap + 8);
      int2 a2v = *(const int2*)(ap + 32);
      int2 a3v = *(const int2*)(ap + 40);
      v8i A;
      A[0] = a0.x;  A[1] = a0.y;  A[2] = a1v.x; A[3] = a1v.y;
      A[4] = a2v.x; A[5] = a2v.y; A[6] = a3v.x; A[7] = a3v.y;

      Cacc = __builtin_amdgcn_wmma_i32_16x16x64_iu8(true, A, true, B, Cacc, false, false);
    }
#pragma unroll
    for (int r = 0; r < 8; ++r)
      s_out[(ot * 16 + hi * 8 + r) * 128 + pb + nn] = (float)Cacc[r] * deq;
  }
  __syncthreads();

  const float* xb = x  + (size_t)n * C_OUT * HW + hw;
  float*       ob = out + (size_t)n * C_OUT * HW + hw;
#pragma unroll
  for (int j = 0; j < 4; ++j) {
    int idx = (j * 256 + tid) * 4;
    int oc = idx >> 7;
    int px = idx & 127;
    float4 v  = *(const float4*)&s_out[oc * 128 + px];
    float4 xr = *(const float4*)(xb + (size_t)oc * HW + px);
    v.x += xr.x; v.y += xr.y; v.z += xr.z; v.w += xr.w;
    *(float4*)(ob + (size_t)oc * HW + px) = v;
  }
}

extern "C" void kernel_launch(void* const* d_in, const int* in_sizes, int n_in,
                              void* d_out, int out_size, void* d_ws, size_t ws_size,
                              hipStream_t stream) {
  const float* x  = (const float*)d_in[0];
  const float* w1 = (const float*)d_in[1];
  const float* w2 = (const float*)d_in[2];
  const float* w3 = (const float*)d_in[3];
  float* out = (float*)d_out;

  char* ws = (char*)d_ws;
  unsigned int* wsh = (unsigned int*)ws;
  signed char* w1q = (signed char*)(ws + OFF_W1Q);
  signed char* w2q = (signed char*)(ws + OFF_W2Q);
  signed char* w3q = (signed char*)(ws + OFF_W3Q);
  short* a1 = (short*)(ws + OFF_A1);
  short* a2 = (short*)(ws + OFF_A2);

  k_init<<<1, 64, 0, stream>>>(wsh);
  k_amax<<<1024, 256, 0, stream>>>((const float4*)x,  NBATCH * C_IN * HW / 4, wsh + 0);
  k_amax<<<6,    256, 0, stream>>>((const float4*)w1, PLANES * C_IN / 4,      wsh + 1);
  k_amax<<<2,    256, 0, stream>>>((const float4*)w2, PLANES * 9 / 4,         wsh + 2);
  k_amax<<<6,    256, 0, stream>>>((const float4*)w3, C_OUT * PLANES / 4,     wsh + 3);
  k_quantw<<<8, 256, 0, stream>>>(w1, w2, w3, wsh, (signed char*)ws);
  k_conv1<<<NPIX / 128, 256, 0, stream>>>(x, w1q, wsh, a1, (int*)wsh + 4);
  k_dw<<<NBATCH * 196, 256, 0, stream>>>(a1, w2q, wsh, a2, (int*)wsh + 5);
  k_conv3<<<NPIX / 128, 256, 0, stream>>>(a2, w3q, wsh, x, out);
}
